// LPGCNEDGNN_51771535786413
// MI455X (gfx1250) — compile-verified, matrix operand order checked by reading, not run on hardware
//
#include <hip/hip_runtime.h>

#define N_NODES 100000
#define N_FEAT  128
#define DIM     64
#define N_EDGES 1600000
#define NNZ_    1000000
#define N_HE    50000

typedef __attribute__((ext_vector_type(16))) __bf16 v16bf;
typedef __attribute__((ext_vector_type(8)))  float  v8f;

union V16 { v16bf v; unsigned u[8]; };

__device__ __forceinline__ unsigned bf16_rne(float f) {
  unsigned u = __float_as_uint(f);
  return (u + 0x7FFFu + ((u >> 16) & 1u)) >> 16;   // round-to-nearest-even
}
__device__ __forceinline__ unsigned pack2(float a, float b) {
  return bf16_rne(a) | (bf16_rne(b) << 16);
}

// ---------------------------------------------------------------------------
// Generic fused GEMM:  C[M x 64] = act( gather(A) [M x K] @ W [K x 64] + bias )
//   * K, K0 compile-time: loader branches resolve statically, WMMA loop fully
//     unrolls (K/32 back-to-back v_wmma per tile, two barriers per tile)
//   * A rows optionally gathered via idx0/idx1 (two K0-split sources = concat);
//     gather indices hoisted to per-tile row pointers (loaded once, not per k)
//   * T row-tiles per block amortize the LDS weight panel staging
//   * output either stored row-major or atomically scattered via `scat` index
// ---------------------------------------------------------------------------
template <int K, int K0, int T>
__global__ __launch_bounds__(128)
void wmma_gemm64(const float* __restrict__ A0, const float* __restrict__ A1,
                 const int* __restrict__ idx0, const int* __restrict__ idx1,
                 const float* __restrict__ W,  const float* __restrict__ bias,
                 float* __restrict__ Cout, const int* __restrict__ scat,
                 int M, int relu)
{
  constexpr int KH  = K / 2;        // packed bf16x2 pairs along K
  constexpr int STR = KH + 2;       // padded row stride (conflict-free reads)

  __shared__ unsigned Wl[KH * DIM]; // (K/2) x 64 packed bf16x2
  __shared__ unsigned At[16 * STR]; // full 16 x K tile, packed pairs

  const int tid  = threadIdx.x;
  const int lane = tid & 31;
  const int wv   = tid >> 5;        // wave id 0..3 -> 16-col tile
  const int hi   = lane >> 4;       // half-wave
  const int mn   = lane & 15;       // M index (A) / N index (B,C)
  const int col0 = wv * 16;
  const int r    = tid >> 3;        // loader row 0..15
  const int lq   = tid & 7;         // loader lane-in-row

  // Stage W once: Wl[kk*64+n] = (W[2kk][n], W[2kk+1][n])
  for (int p = tid; p < KH * DIM; p += 128) {
    int kk = p >> 6, n = p & 63;
    Wl[p] = pack2(W[(2 * kk) * DIM + n], W[(2 * kk + 1) * DIM + n]);
  }

  const int tileBase = blockIdx.x * (16 * T);

#pragma unroll 1
  for (int t = 0; t < T; ++t) {
    const int row0 = tileBase + t * 16;
    const int gi   = row0 + r;

    // Hoisted per-tile row pointers (gather index loaded once per tile)
    const float* p0 = nullptr;
    const float* p1 = nullptr;
    if (gi < M) {
      int ri0 = idx0 ? idx0[gi] : gi;
      p0 = A0 + (size_t)ri0 * K0;
      if (K > K0) {
        int ri1 = idx1 ? idx1[gi] : gi;
        p1 = A1 + (size_t)ri1 * (K - K0);
      }
    }

    __syncthreads();  // protects W (t==0) and previous tile's At readers

    // Stage the full 16 x K A tile (float4 loads, static concat split)
#pragma unroll
    for (int q = lq; q < K / 4; q += 8) {
      int col = q * 4;
      float4 val = make_float4(0.f, 0.f, 0.f, 0.f);
      if (gi < M)
        val = *(const float4*)((col < K0) ? (p0 + col) : (p1 + (col - K0)));
      At[r * STR + q * 2 + 0] = pack2(val.x, val.y);
      At[r * STR + q * 2 + 1] = pack2(val.z, val.w);
    }
    __syncthreads();

    // K/32 back-to-back WMMAs straight from LDS (ISA 7.12.2 16-bit layouts)
    v8f acc = {0.f, 0.f, 0.f, 0.f, 0.f, 0.f, 0.f, 0.f};
#pragma unroll
    for (int k0 = 0; k0 < K; k0 += 32) {
      V16 a, b;
#pragma unroll
      for (int v = 0; v < 8; ++v) {
        int kb = ((v & 3) << 1) + (hi << 3) + ((v >> 2) << 4);  // K-pair base
        a.u[v] = At[mn * STR + ((k0 + kb) >> 1)];
        b.u[v] = Wl[((k0 + kb) >> 1) * DIM + col0 + mn];
      }
      acc = __builtin_amdgcn_wmma_f32_16x16x32_bf16(false, a.v, false, b.v,
                                                    (short)0, acc, false, false);
    }

    // Epilogue: bias + relu, then store or atomic segment-scatter
    const int col = col0 + mn;
    const float bb = bias ? bias[col] : 0.f;
#pragma unroll
    for (int v = 0; v < 8; ++v) {
      int go = row0 + v + hi * 8;              // C layout: M = v (+8 hi half)
      if (go < M) {
        float val = acc[v] + bb;
        if (relu) val = fmaxf(val, 0.f);
        if (scat) atomicAdd(&Cout[(size_t)scat[go] * DIM + col], val);
        else      Cout[(size_t)go * DIM + col] = val;
      }
    }
  }
}

// ---------------------------------------------------------------------------
// Elementwise / graph kernels (L2-resident tables -> f32 atomics)
// ---------------------------------------------------------------------------
__global__ void fill1_k(float* p, int n) {
  int i = blockIdx.x * blockDim.x + threadIdx.x;
  if (i < n) p[i] = 1.0f;
}

__global__ void deg_acc_k(const int* __restrict__ dst, float* deg, int e) {
  int i = blockIdx.x * blockDim.x + threadIdx.x;
  if (i < e) atomicAdd(&deg[dst[i]], 1.0f);
}

__global__ void rsqrt_k(float* p, int n) {
  int i = blockIdx.x * blockDim.x + threadIdx.x;
  if (i < n) p[i] = rsqrtf(p[i]);
}

// agg[i,:] = xw[i,:] * dinv[i]^2  (self-loop contribution, also initializes agg)
__global__ void self_init_k(const float* __restrict__ xw,
                            const float* __restrict__ dinv,
                            float* __restrict__ agg, int n) {
  int t = blockIdx.x * blockDim.x + threadIdx.x;
  if (t < n * 16) {
    int i = t >> 4, g = t & 15;
    float s = dinv[i] * dinv[i];
    float4 v = ((const float4*)(xw + (size_t)i * DIM))[g];
    v.x *= s; v.y *= s; v.z *= s; v.w *= s;
    ((float4*)(agg + (size_t)i * DIM))[g] = v;
  }
}

__global__ void edge_scat_k(const int* __restrict__ src, const int* __restrict__ dst,
                            const float* __restrict__ dinv,
                            const float* __restrict__ xw, float* __restrict__ agg, int e) {
  int t = blockIdx.x * blockDim.x + threadIdx.x;
  if (t < e * 16) {
    int ed = t >> 4, g = t & 15;
    int s = src[ed], d = dst[ed];
    float nm = dinv[s] * dinv[d];
    float4 v = ((const float4*)(xw + (size_t)s * DIM))[g];
    float* o = agg + (size_t)d * DIM + g * 4;
    atomicAdd(o + 0, v.x * nm); atomicAdd(o + 1, v.y * nm);
    atomicAdd(o + 2, v.z * nm); atomicAdd(o + 3, v.w * nm);
  }
}

__global__ void bias_act_k(const float* __restrict__ agg, const float* __restrict__ b,
                           float* __restrict__ out, int n, int relu) {
  int t = blockIdx.x * blockDim.x + threadIdx.x;
  if (t < n * DIM) {
    float v = agg[t] + b[t & 63];
    if (relu) v = fmaxf(v, 0.f);
    out[t] = v;
  }
}

__global__ void counts_k(const int* __restrict__ hv, const int* __restrict__ he,
                         float* cv, float* ce, int nnz) {
  int i = blockIdx.x * blockDim.x + threadIdx.x;
  if (i < nnz) { atomicAdd(&cv[hv[i]], 1.0f); atomicAdd(&ce[he[i]], 1.0f); }
}

__global__ void xe_scat_k(const int* __restrict__ hv, const int* __restrict__ he,
                          const float* __restrict__ hw1, float* __restrict__ xe, int nnz) {
  int t = blockIdx.x * blockDim.x + threadIdx.x;
  if (t < nnz * 16) {
    int i = t >> 4, g = t & 15;
    float4 v = ((const float4*)(hw1 + (size_t)hv[i] * DIM))[g];
    float* o = xe + (size_t)he[i] * DIM + g * 4;
    atomicAdd(o + 0, v.x); atomicAdd(o + 1, v.y);
    atomicAdd(o + 2, v.z); atomicAdd(o + 3, v.w);
  }
}

__global__ void xe_norm_k(float* __restrict__ xe, const float* __restrict__ ce, int nhe) {
  int t = blockIdx.x * blockDim.x + threadIdx.x;
  if (t < nhe * DIM) xe[t] *= 1.0f / fmaxf(ce[t >> 6], 1.0f);
}

// h2 = 0.5*(Xv_sum/max(cnt,1)) + 0.5*h0
__global__ void combine_k(const float* __restrict__ xv, const float* __restrict__ cv,
                          const float* __restrict__ h0, float* __restrict__ out, int n) {
  int t = blockIdx.x * blockDim.x + threadIdx.x;
  if (t < n * DIM) {
    float inv = 1.0f / fmaxf(cv[t >> 6], 1.0f);
    out[t] = 0.5f * xv[t] * inv + 0.5f * h0[t];
  }
}

__global__ void head_dot_k(const float* __restrict__ fused, const float* __restrict__ w,
                           const float* __restrict__ b, float* __restrict__ out, int n) {
  int i = blockIdx.x * blockDim.x + threadIdx.x;
  if (i < n) {
    const float4* f = (const float4*)(fused + (size_t)i * DIM);
    const float4* ww = (const float4*)w;
    float s = 0.f;
#pragma unroll
    for (int g = 0; g < 16; ++g) {
      float4 a = f[g], c = ww[g];
      s += a.x * c.x + a.y * c.y + a.z * c.z + a.w * c.w;
    }
    out[i] = s + b[0];
  }
}

// ---------------------------------------------------------------------------
extern "C" void kernel_launch(void* const* d_in, const int* in_sizes, int n_in,
                              void* d_out, int out_size, void* d_ws, size_t ws_size,
                              hipStream_t stream) {
  (void)in_sizes; (void)n_in; (void)out_size; (void)ws_size;

  const float* x       = (const float*)d_in[0];
  const int*   ei      = (const int*)  d_in[1];
  const int*   hv      = (const int*)  d_in[2];
  const int*   he      = (const int*)  d_in[3];
  const float* conv1_W = (const float*)d_in[5];
  const float* conv1_b = (const float*)d_in[6];
  const float* conv2_W = (const float*)d_in[7];
  const float* conv2_b = (const float*)d_in[8];
  const float* lin_in_W= (const float*)d_in[9];
  const float* lin_in_b= (const float*)d_in[10];
  const float* W1      = (const float*)d_in[11];
  const float* b1      = (const float*)d_in[12];
  const float* W2      = (const float*)d_in[13];
  const float* b2      = (const float*)d_in[14];
  const float* Wu      = (const float*)d_in[15];
  const float* bu      = (const float*)d_in[16];
  const float* cls_W   = (const float*)d_in[17];
  const float* cls_b   = (const float*)d_in[18];
  const float* lp_W    = (const float*)d_in[19];
  const float* lp_b    = (const float*)d_in[20];
  const float* head_W  = (const float*)d_in[21];
  const float* head_b  = (const float*)d_in[22];
  float* out = (float*)d_out;

  // Workspace layout (floats)
  float* ws = (float*)d_ws;
  const size_t NF = (size_t)N_NODES * DIM;
  float* bufA = ws;                    // scratch GEMM out / xw / hw1 / x_hyper
  float* bufB = ws + NF;               // aggregation targets / fused
  float* bufC = ws + 2 * NF;           // x1 / x_gnn
  float* bufD = ws + 3 * NF;           // h / h3
  float* Xe   = ws + 4 * NF;           // N_HE x 64
  float* cE   = Xe + (size_t)N_HE * DIM;
  float* cV   = cE + N_HE;
  float* dinv = cV + N_NODES;

  const int* esrc = ei;                // edge_index[0]
  const int* edst = ei + N_EDGES;      // edge_index[1]

  dim3 B(256);
  auto gb = [](int n) { return dim3((unsigned)((n + 255) / 256)); };
  auto gg = [](int m) { return dim3((unsigned)((m + 63) / 64)); };  // 4 tiles/block
  dim3 GB(128);

  // ---------------- GCN branch ----------------
  // xw1 = x @ conv1_W
  wmma_gemm64<128, 128, 4><<<gg(N_NODES), GB, 0, stream>>>(
      x, nullptr, nullptr, nullptr, conv1_W, nullptr, bufA, nullptr, N_NODES, 0);
  // deg (incl. self loop) -> dinv
  fill1_k<<<gb(N_NODES), B, 0, stream>>>(dinv, N_NODES);
  deg_acc_k<<<gb(N_EDGES), B, 0, stream>>>(edst, dinv, N_EDGES);
  rsqrt_k<<<gb(N_NODES), B, 0, stream>>>(dinv, N_NODES);
  // conv1 aggregate: self-loop init + edge scatter, then bias + relu
  self_init_k<<<gb(N_NODES * 16), B, 0, stream>>>(bufA, dinv, bufB, N_NODES);
  edge_scat_k<<<gb(N_EDGES * 16), B, 0, stream>>>(esrc, edst, dinv, bufA, bufB, N_EDGES);
  bias_act_k<<<gb(N_NODES * DIM), B, 0, stream>>>(bufB, conv1_b, bufC, N_NODES, 1);
  // conv2
  wmma_gemm64<64, 64, 4><<<gg(N_NODES), GB, 0, stream>>>(
      bufC, nullptr, nullptr, nullptr, conv2_W, nullptr, bufA, nullptr, N_NODES, 0);
  self_init_k<<<gb(N_NODES * 16), B, 0, stream>>>(bufA, dinv, bufB, N_NODES);
  edge_scat_k<<<gb(N_EDGES * 16), B, 0, stream>>>(esrc, edst, dinv, bufA, bufB, N_EDGES);
  bias_act_k<<<gb(N_NODES * DIM), B, 0, stream>>>(bufB, conv2_b, bufC, N_NODES, 0); // x_gnn

  // ---------------- Hypergraph branch ----------------
  // h = relu(x @ lin_in_W + b)
  wmma_gemm64<128, 128, 4><<<gg(N_NODES), GB, 0, stream>>>(
      x, nullptr, nullptr, nullptr, lin_in_W, lin_in_b, bufD, nullptr, N_NODES, 1);
  // hw1 = h @ W1 + b1
  wmma_gemm64<64, 64, 4><<<gg(N_NODES), GB, 0, stream>>>(
      bufD, nullptr, nullptr, nullptr, W1, b1, bufA, nullptr, N_NODES, 0);
  hipMemsetAsync(Xe, 0, (size_t)N_HE * DIM * sizeof(float), stream);
  hipMemsetAsync(cE, 0, (size_t)N_HE * sizeof(float), stream);
  hipMemsetAsync(cV, 0, (size_t)N_NODES * sizeof(float), stream);
  hipMemsetAsync(bufB, 0, NF * sizeof(float), stream);   // Xv_sum
  counts_k<<<gb(NNZ_), B, 0, stream>>>(hv, he, cV, cE, NNZ_);
  xe_scat_k<<<gb(NNZ_ * 16), B, 0, stream>>>(hv, he, bufA, Xe, NNZ_);
  xe_norm_k<<<gb(N_HE * DIM), B, 0, stream>>>(Xe, cE, N_HE);
  // m = relu(concat(h[hv], Xe[he]) @ W2 + b2), scatter-summed into Xv_sum by hv
  wmma_gemm64<128, 64, 4><<<gg(NNZ_), GB, 0, stream>>>(
      bufD, Xe, hv, he, W2, b2, bufB, hv, NNZ_, 1);
  // h2 = 0.5*Xv + 0.5*h0
  combine_k<<<gb(N_NODES * DIM), B, 0, stream>>>(bufB, cV, bufD, bufA, N_NODES);
  // h3 = relu(h2 @ Wu + bu)
  wmma_gemm64<64, 64, 4><<<gg(N_NODES), GB, 0, stream>>>(
      bufA, nullptr, nullptr, nullptr, Wu, bu, bufD, nullptr, N_NODES, 1);
  // x_hyper = h3 @ cls_W + cls_b
  wmma_gemm64<64, 64, 4><<<gg(N_NODES), GB, 0, stream>>>(
      bufD, nullptr, nullptr, nullptr, cls_W, cls_b, bufA, nullptr, N_NODES, 0);

  // ---------------- Fuse + head ----------------
  // fused = concat(x_gnn, x_hyper) @ lp_W + lp_b   (identity two-source concat)
  wmma_gemm64<128, 64, 4><<<gg(N_NODES), GB, 0, stream>>>(
      bufC, bufA, nullptr, nullptr, lp_W, lp_b, bufB, nullptr, N_NODES, 0);
  head_dot_k<<<gb(N_NODES), B, 0, stream>>>(bufB, head_W, head_b, out, N_NODES);
}
